// my_pooling_5617817223818
// MI455X (gfx1250) — matrix operand order repaired and checked by reading, not compiled, not run on hardware
//
#include <hip/hip_runtime.h>
#include <hip/hip_bf16.h>

// ---------------------------------------------------------------------------
// MI455X (gfx1250) implementation.
//  - GEMM core: V_WMMA_F32_16X16X32_F16, one 16x16 C tile per wave32.
//  - gemm12 fuses x@w1 and x@w2 (shared A fragment -> 2 WMMAs/step) plus the
//    SiLU(x1)*x2 epilogue, writing f16 H directly.
//  - gemm3 does H@w3+b3 with N-bound guards (handles final out_d == 1).
//  - LayerNorm fused with f32->f16 conversion for the next layer's A operand.
//  - SiLU sigmoid uses v_rcp_f32 (hardware reciprocal) instead of the IEEE
//    divide expansion -- removes the v_div_scale/v_div_fmas chains from the
//    dominant epilogue.
// ---------------------------------------------------------------------------

typedef __attribute__((ext_vector_type(16))) _Float16 v16h;
typedef __attribute__((ext_vector_type(8)))  _Float16 v8h;
typedef __attribute__((ext_vector_type(8)))  float    v8f;

// A-fragment (16-bit A 16x32, wave32) per CDNA5 ISA layout:
//  lanes 0-15 : row = lane,    K = {k0+0..7, k0+16..23}
//  lanes 16-31: row = lane-16, K = {k0+8..15, k0+24..31}
__device__ inline v16h load_a_frag(const _Float16* __restrict__ X, int K,
                                   int m0, int k0, int lane) {
  const int sh = (lane >= 16) ? 8 : 0;
  const _Float16* arow = X + (size_t)(m0 + (lane & 15)) * K + k0 + sh;
  v8h lo = *(const v8h*)(arow);        // 16B load
  v8h hi = *(const v8h*)(arow + 16);   // 16B load
  v16h a;
#pragma unroll
  for (int i = 0; i < 8; ++i) { a[i] = lo[i]; a[i + 8] = hi[i]; }
  return a;
}

// B-fragment (16-bit B 32x16): lane l holds column N = n0+l%16, K = {k0+8h+0..7, k0+16+8h+0..7},
// converted f32 -> f16 in-register (v_cvt_pk_f16_f32 pairs).
__device__ inline v16h load_b_frag_f32(const float* __restrict__ W, int N,
                                       int k0, int n0, int lane) {
  const float* wrow = W + (size_t)(k0 + ((lane >> 4) << 3)) * N + n0 + (lane & 15);
  v16h b;
#pragma unroll
  for (int i = 0; i < 16; ++i) b[i] = (_Float16)wrow[(size_t)((i & 7) + ((i >> 3) << 4)) * N];
  return b;
}

// silu(x) = x * sigmoid(x); sigmoid via v_exp_f32 + v_rcp_f32 (arg >= 1, so
// the 1-ULP hardware reciprocal is well conditioned).
__device__ inline float silu_fast(float x) {
  return x * __builtin_amdgcn_rcpf(1.0f + __expf(-x));
}

// H = silu(X@W1 + b1) * (X@W2 + b2), f16 output.  block = (32,4) waves,
// grid = (M/16, N/64).  Two WMMAs per K-step share one A fragment.
__global__ __launch_bounds__(128)
void gemm12_silu_kernel(const _Float16* __restrict__ X,
                        const float* __restrict__ W1, const float* __restrict__ B1,
                        const float* __restrict__ W2, const float* __restrict__ B2,
                        _Float16* __restrict__ H, int M, int K, int N) {
  const int lane = threadIdx.x;
  const int m0 = blockIdx.x * 16;
  const int n0 = (blockIdx.y * 4 + threadIdx.y) * 16;
  v8f acc1 = {}; v8f acc2 = {};
  for (int k0 = 0; k0 < K; k0 += 32) {
    v16h a  = load_a_frag(X, K, m0, k0, lane);
    v16h b1 = load_b_frag_f32(W1, N, k0, n0, lane);
    v16h b2 = load_b_frag_f32(W2, N, k0, n0, lane);
    acc1 = __builtin_amdgcn_wmma_f32_16x16x32_f16(false, a, false, b1,
                                                  (short)0, acc1, false, false);
    acc2 = __builtin_amdgcn_wmma_f32_16x16x32_f16(false, a, false, b2,
                                                  (short)0, acc2, false, false);
  }
  // D layout: lane<16 -> (M = m0+r,   N = lane); lane>=16 -> (M = m0+8+r, N = lane-16)
  const int nloc = n0 + (lane & 15);
  const float bb1 = B1[nloc], bb2 = B2[nloc];
  const int mb = m0 + ((lane < 16) ? 0 : 8);
#pragma unroll
  for (int r = 0; r < 8; ++r) {
    float x1 = acc1[r] + bb1;
    float x2 = acc2[r] + bb2;
    float h = silu_fast(x1) * x2;
    H[(size_t)(mb + r) * N + nloc] = (_Float16)h;
  }
}

// O = H@W3 + b3, f32 output.  Handles N not multiple of 16 (final out_d = 1).
__global__ __launch_bounds__(32)
void gemm3_kernel(const _Float16* __restrict__ Hh, const float* __restrict__ W3,
                  const float* __restrict__ B3, float* __restrict__ O,
                  int M, int K, int N) {
  const int lane = threadIdx.x;
  const int m0 = blockIdx.x * 16;
  const int n0 = blockIdx.y * 16;
  v8f acc = {};
  for (int k0 = 0; k0 < K; k0 += 32) {
    v16h a = load_a_frag(Hh, K, m0, k0, lane);
    const float* wrow = W3 + (size_t)(k0 + ((lane >> 4) << 3)) * N;
    v16h b;
#pragma unroll
    for (int i = 0; i < 16; ++i) {
      int n = n0 + (lane & 15);
      b[i] = (_Float16)((n < N) ? wrow[(size_t)((i & 7) + ((i >> 3) << 4)) * N + n] : 0.0f);
    }
    acc = __builtin_amdgcn_wmma_f32_16x16x32_f16(false, a, false, b,
                                                 (short)0, acc, false, false);
  }
  const int nloc = n0 + (lane & 15);
  if (nloc < N) {
    const float bb = B3[nloc];
    const int mb = m0 + ((lane < 16) ? 0 : 8);
#pragma unroll
    for (int r = 0; r < 8; ++r)
      O[(size_t)(mb + r) * N + nloc] = acc[r] + bb;
  }
}

// Per-row LayerNorm over N (<=384), fused with f32->f16 conversion into the
// next layer's A buffer.  One block per row.
__global__ __launch_bounds__(128)
void layernorm_to_f16_kernel(const float* __restrict__ O,
                             const float* __restrict__ G,
                             const float* __restrict__ Bt,
                             _Float16* __restrict__ X, int N) {
  __shared__ float red[128];
  const int row = blockIdx.x;
  const int t = threadIdx.x;
  const float* o = O + (size_t)row * N;
  float v[3] = {0.f, 0.f, 0.f};
  float s = 0.f;
  for (int j = t, i = 0; j < N; j += 128, ++i) { v[i] = o[j]; s += v[i]; }
  red[t] = s; __syncthreads();
  for (int off = 64; off > 0; off >>= 1) { if (t < off) red[t] += red[t + off]; __syncthreads(); }
  const float mu = red[0] / (float)N;
  __syncthreads();
  float sq = 0.f;
  for (int j = t, i = 0; j < N; j += 128, ++i) { float d = v[i] - mu; sq += d * d; }
  red[t] = sq; __syncthreads();
  for (int off = 64; off > 0; off >>= 1) { if (t < off) red[t] += red[t + off]; __syncthreads(); }
  const float rstd = rsqrtf(red[0] / (float)N + 1e-5f);
  _Float16* x = X + (size_t)row * N;
  for (int j = t, i = 0; j < N; j += 128, ++i)
    x[j] = (_Float16)((v[i] - mu) * rstd * G[j] + Bt[j]);
}

// Masked mean-pool + L2 norm; also emits lengths and zeroes mod accumulator.
__global__ __launch_bounds__(256)
void pool_kernel(const float* __restrict__ tok, const int* __restrict__ mask,
                 int* __restrict__ lengths, float* __restrict__ mean_emb,
                 float* __restrict__ mod, int B, int S, int D) {
  __shared__ float msk[1024];
  __shared__ float red[256];
  const int b = blockIdx.x;
  const int t = threadIdx.x;
  int lsum = 0;
  for (int s = t; s < S; s += 256) { int m = mask[b * S + s]; msk[s] = (float)m; lsum += m; }
  red[t] = (float)lsum; __syncthreads();
  for (int off = 128; off > 0; off >>= 1) { if (t < off) red[t] += red[t + off]; __syncthreads(); }
  const float len = red[0];
  if (t == 0) lengths[b] = (int)len;
  __syncthreads();
  const float denom = fmaxf(len, 1e-9f);
  const float* tb = tok + (size_t)b * S * D;
  float m3[3];
#pragma unroll
  for (int i = 0; i < 3; ++i) {
    const int d = t + i * 256;           // D = 768 = 3 * 256
    float acc = 0.f;
    for (int s = 0; s < S; ++s) acc += tb[(size_t)s * D + d] * msk[s];
    m3[i] = acc / denom;
  }
  float sq = m3[0] * m3[0] + m3[1] * m3[1] + m3[2] * m3[2];
  red[t] = sq; __syncthreads();
  for (int off = 128; off > 0; off >>= 1) { if (t < off) red[t] += red[t + off]; __syncthreads(); }
  const float inv = 1.0f / fmaxf(sqrtf(red[0]), 1e-12f);
#pragma unroll
  for (int i = 0; i < 3; ++i) {
    const int d = t + i * 256;
    mean_emb[(size_t)b * D + d] = m3[i] * inv;
    mod[(size_t)b * D + d] = 0.f;
  }
}

// Ragged linear resize to `size`, transposed to (B, D, size), f16 row-major
// (K = size contiguous -> valid WMMA A operand).  One block per (b,d) row.
__global__ __launch_bounds__(128)
void resize_kernel(const float* __restrict__ tok, const int* __restrict__ lengths,
                   _Float16* __restrict__ X, int B, int S, int D, int size) {
  const int bd = blockIdx.x;            // b*D + d
  const int b = bd / D;
  const int d = bd - b * D;
  const int L = lengths[b];
  const float scale = (float)L / (float)size;
  const float* tb = tok + (size_t)b * S * D + d;
  _Float16* xrow = X + (size_t)bd * size;
  for (int s = threadIdx.x; s < size; s += 128) {
    float x = fmaxf(((float)s + 0.5f) * scale - 0.5f, 0.f);
    int i0 = min((int)floorf(x), L - 1);
    int i1 = min(i0 + 1, L - 1);
    float w = x - (float)i0;
    float v = tb[(size_t)i0 * D] * (1.f - w) + tb[(size_t)i1 * D] * w;
    xrow[s] = (_Float16)v;
  }
}

// mod += O[:,0] * amp*exp(-(L-center)^2/(2*var))   (per batch Gaussian)
__global__ __launch_bounds__(256)
void accum_kernel(const float* __restrict__ O, const int* __restrict__ lengths,
                  const float* __restrict__ amp, const float* __restrict__ var,
                  float* __restrict__ mod, int D, float center, int M) {
  int m = blockIdx.x * 256 + threadIdx.x;
  if (m >= M) return;
  int b = m / D;
  float dd = (float)lengths[b] - center;
  float g = amp[0] * __expf(-dd * dd * __builtin_amdgcn_rcpf(2.0f * var[0]));
  mod[m] += O[m] * g;
}

// out = l2norm(mean_emb + mod)
__global__ __launch_bounds__(256)
void final_kernel(const float* __restrict__ mean_emb, const float* __restrict__ mod,
                  float* __restrict__ out, int D) {
  __shared__ float red[256];
  const int b = blockIdx.x;
  const int t = threadIdx.x;
  float v3[3]; float sq = 0.f;
#pragma unroll
  for (int i = 0; i < 3; ++i) {
    int d = t + i * 256;
    float v = mean_emb[(size_t)b * D + d] + mod[(size_t)b * D + d];
    v3[i] = v; sq += v * v;
  }
  red[t] = sq; __syncthreads();
  for (int off = 128; off > 0; off >>= 1) { if (t < off) red[t] += red[t + off]; __syncthreads(); }
  const float inv = 1.0f / fmaxf(sqrtf(red[0]), 1e-12f);
#pragma unroll
  for (int i = 0; i < 3; ++i) {
    int d = t + i * 256;
    out[(size_t)b * D + d] = v3[i] * inv;
  }
}

// ---------------------------------------------------------------------------
// Launch.  d_in follows JAX pytree flattening (dict keys sorted):
//   [0] token_embeddings  [1] attention_mask
//   [2..9]  gauss_{128,256,384,512}.{amp,var}
//   then net_128, net_256, net_384, net_512; per layer (sorted keys):
//   b1,b2,b3,[ln_b,ln_g],w1,w2,w3   (final layer of each net has no LN)
// ---------------------------------------------------------------------------
extern "C" void kernel_launch(void* const* d_in, const int* in_sizes, int n_in,
                              void* d_out, int out_size, void* d_ws, size_t ws_size,
                              hipStream_t stream) {
  (void)in_sizes; (void)n_in; (void)out_size; (void)ws_size;
  const int B = 32, S = 1024, D = 768;
  const int M = B * D;                      // 24576 GEMM rows

  const float* tok = (const float*)d_in[0];
  const int* mask = (const int*)d_in[1];

  // Workspace carve (256B aligned), ~82 MB total.
  char* ws = (char*)d_ws;
  size_t off = 0;
  auto carve = [&](size_t bytes) -> void* {
    void* p = ws + off;
    off += (bytes + 255) & ~(size_t)255;
    return p;
  };
  int*      lengths  = (int*)      carve((size_t)B * sizeof(int));
  float*    mean_emb = (float*)    carve((size_t)M * sizeof(float));
  float*    mod      = (float*)    carve((size_t)M * sizeof(float));
  _Float16* bufX     = (_Float16*) carve((size_t)M * 512 * sizeof(_Float16));
  _Float16* bufH     = (_Float16*) carve((size_t)M * 384 * sizeof(_Float16));
  float*    bufO     = (float*)    carve((size_t)M * 384 * sizeof(float));

  pool_kernel<<<B, 256, 0, stream>>>(tok, mask, lengths, mean_emb, mod, B, S, D);

  const int sizes_sorted[4] = {128, 256, 384, 512};   // matches sorted key order
  const int gauss_base[4]  = {2, 4, 6, 8};
  int idx = 10;                                        // first net_128 array
  for (int ni = 0; ni < 4; ++ni) {
    const int size = sizes_sorted[ni];
    const int nlay = size / 128 + 1;
    auto dim = [&](int j) { int v = size - 128 * j; return v > 0 ? v : 64; };

    resize_kernel<<<B * D, 128, 0, stream>>>(tok, lengths, bufX, B, S, D, size);

    for (int li = 0; li < nlay; ++li) {
      const bool last = (li == nlay - 1);
      int in_d, hid_d, out_d; bool ln;
      if (!last) { in_d = dim(li); hid_d = dim(li + 1); out_d = hid_d; ln = true; }
      else       { in_d = 64;      hid_d = 64;          out_d = 1;     ln = false; }

      const float* b1 = (const float*)d_in[idx + 0];
      const float* b2 = (const float*)d_in[idx + 1];
      const float* b3 = (const float*)d_in[idx + 2];
      const float *lnb = nullptr, *lng = nullptr, *w1, *w2, *w3;
      if (ln) {
        lnb = (const float*)d_in[idx + 3];
        lng = (const float*)d_in[idx + 4];
        w1  = (const float*)d_in[idx + 5];
        w2  = (const float*)d_in[idx + 6];
        w3  = (const float*)d_in[idx + 7];
        idx += 8;
      } else {
        w1  = (const float*)d_in[idx + 3];
        w2  = (const float*)d_in[idx + 4];
        w3  = (const float*)d_in[idx + 5];
        idx += 6;
      }

      dim3 g12(M / 16, hid_d / 64), blk12(32, 4);
      gemm12_silu_kernel<<<g12, blk12, 0, stream>>>(bufX, w1, b1, w2, b2, bufH,
                                                    M, in_d, hid_d);
      dim3 g3(M / 16, (out_d + 15) / 16);
      gemm3_kernel<<<g3, 32, 0, stream>>>(bufH, w3, b3, bufO, M, hid_d, out_d);
      if (ln)
        layernorm_to_f16_kernel<<<M, 128, 0, stream>>>(bufO, lng, lnb, bufX, out_d);
    }

    const float* amp = (const float*)d_in[gauss_base[ni]];
    const float* var = (const float*)d_in[gauss_base[ni] + 1];
    accum_kernel<<<(M + 255) / 256, 256, 0, stream>>>(bufO, lengths, amp, var,
                                                      mod, D, (float)size, M);
  }

  final_kernel<<<B, 256, 0, stream>>>(mean_emb, mod, (float*)d_out, D);
}
